// WindowAttention_3d_63634235457932
// MI455X (gfx1250) — matrix operand order repaired and checked by reading, not compiled, not run on hardware
//
#include <hip/hip_runtime.h>
#include <hip/hip_bf16.h>
#include <stdint.h>

// ---------------- problem constants ----------------
#define NUM_HEADS 8
#define DIM       256
#define SIZE      64      // tokens per window
#define HD        32      // head dim
#define QKV_N     768     // 3*DIM

// ---------------- WMMA types ----------------
typedef __attribute__((ext_vector_type(16))) __bf16 v16bf;
typedef __attribute__((ext_vector_type(8)))  float  v8f;

// ---------------- LDS geometry (elements, bf16 = ushort) ----------------
#define XLD 264           // x-tile / q / k / attn_out row stride (256 + 8 pad)
#define QLD 264
#define VLD 72            // v-transposed row stride (64 + 8 pad)
#define PLD 72            // probs row stride

// byte layout of dynamic LDS
#define REGA_BYTES (NUM_HEADS * SIZE * PLD * 2)    // probs (73728 B); overlays x-tile (33792 B)
#define Q_OFF      (REGA_BYTES)
#define K_OFF      (Q_OFF + SIZE * QLD * 2)
#define VT_OFF     (K_OFF + SIZE * QLD * 2)
#define SMEM_BYTES (VT_OFF + DIM * VLD * 2)        // 178176 bytes

// ---------------- helpers ----------------
__device__ __forceinline__ unsigned short f2bfu(float x) {
  union { float f; unsigned u; } c; c.f = x;
  unsigned r = c.u + 0x7FFFu + ((c.u >> 16) & 1u);   // round-to-nearest-even
  return (unsigned short)(r >> 16);
}

union FragBF { v16bf v; uint4 q[2]; };

// A-matrix 16x32 bf16 fragment from row-major [16 x >=32] (ISA 7.12.2):
// lanes 0-15: row M=lane, elems 0-7 = K 0..7, elems 8-15 = K 16..23
// lanes 16-31: row M=lane-16, K 8..15 and K 24..31
__device__ __forceinline__ v16bf load_a(const unsigned short* p, int ld, int lane) {
  const int r  = lane & 15;
  const int kb = (lane >> 4) << 3;
  const unsigned short* b = p + r * ld + kb;
  FragBF f;
  f.q[0] = *(const uint4*)(b);
  f.q[1] = *(const uint4*)(b + 16);
  return f.v;
}

// B-matrix 32x16 bf16 fragment from row-major B^T [16 x >=32]:
// lanes 0-15: col N=lane, elems 0-7 = K 0..7, elems 8-15 = K 16..23 ; lanes 16-31: N=lane-16, K 8..15 and K 24..31
__device__ __forceinline__ v16bf load_b(const unsigned short* p, int ld, int lane) {
  const int n  = lane & 15;
  const int kb = (lane >> 4) << 3;
  const unsigned short* b = p + n * ld + kb;
  FragBF f;
  f.q[0] = *(const uint4*)(b);
  f.q[1] = *(const uint4*)(b + 16);
  return f.v;
}

__device__ __forceinline__ v8f wmma_bf16(v16bf a, v16bf b, v8f c) {
  return __builtin_amdgcn_wmma_f32_16x16x32_bf16(false, a, false, b, (short)0, c,
                                                 false, false);
}

// ---------------- prep kernels (weights -> bf16 transposed, bias table gather) ----------------
// wt[n*K + k] = bf16(w[k*N + n])   (w is [K x N] row-major, wt is [N x K] row-major)
__global__ void prep_wT(const float* __restrict__ w, unsigned short* __restrict__ wt,
                        int K, int N) {
  int idx = blockIdx.x * blockDim.x + threadIdx.x;
  if (idx >= K * N) return;
  int n = idx / K;
  int k = idx - n * K;
  wt[idx] = f2bfu(w[(size_t)k * N + n]);
}

// rb[h*4096 + qi*64 + kj] = table[REL_IDX(qi,kj)*8 + h]
__global__ void prep_relbias(const float* __restrict__ table, float* __restrict__ rb) {
  int idx = blockIdx.x * blockDim.x + threadIdx.x;
  if (idx >= NUM_HEADS * SIZE * SIZE) return;
  int h  = idx >> 12;
  int qi = (idx >> 6) & 63;
  int kj = idx & 63;
  int dt = (qi >> 4)       - (kj >> 4)       + 3;
  int dh = ((qi >> 2) & 3) - ((kj >> 2) & 3) + 3;
  int dw = (qi & 3)        - (kj & 3)        + 3;
  int r  = dt * 49 + dh * 7 + dw;
  rb[idx] = table[r * NUM_HEADS + h];
}

// ---------------- fused per-window attention kernel ----------------
__global__ void __launch_bounds__(256, 1)
win_attn_fused(const float* __restrict__ x,
               const float* __restrict__ qkv_b,
               const float* __restrict__ proj_b,
               const unsigned short* __restrict__ wqkvT,   // [768 x 256] bf16
               const unsigned short* __restrict__ projT,   // [256 x 256] bf16
               const float* __restrict__ relb,             // [8 x 64 x 64]
               float* __restrict__ out) {
  extern __shared__ __align__(16) char smem[];
  unsigned short* xt    = (unsigned short*)(smem);          // [64][XLD]  (phase 1-2)
  unsigned short* probs = (unsigned short*)(smem);          // [8][64][PLD] (phase 3)
  unsigned short* qs    = (unsigned short*)(smem + Q_OFF);  // [64][QLD]; later attn_out
  unsigned short* ks    = (unsigned short*)(smem + K_OFF);  // [64][QLD]
  unsigned short* vt    = (unsigned short*)(smem + VT_OFF); // [256][VLD]  v transposed

  const int tid  = threadIdx.x;
  const int lane = tid & 31;
  const int wave = tid >> 5;
  const int win  = blockIdx.x;
  const int nn   = lane & 15;
  const int hi8  = (lane >> 4) << 3;

  // ---- phase 1: load x window tile, convert to bf16 in LDS ----
  {
    const float4* xs = (const float4*)(x + (size_t)win * SIZE * DIM);
    #pragma unroll
    for (int it = 0; it < 16; ++it) {
      int i = it * 256 + tid;          // float4 index in [0, 4096)
      float4 f = xs[i];
      int e   = i << 2;
      int tok = e >> 8;
      int col = e & 255;
      unsigned p0 = (unsigned)f2bfu(f.x) | ((unsigned)f2bfu(f.y) << 16);
      unsigned p1 = (unsigned)f2bfu(f.z) | ((unsigned)f2bfu(f.w) << 16);
      unsigned* d = (unsigned*)(xt + tok * XLD + col);
      d[0] = p0; d[1] = p1;
    }
  }
  __syncthreads();

  // ---- phase 2: QKV GEMM  [64x256] @ W[256x768] -> q,k (row-major), v (transposed) ----
  const float scale = 0.17677669529663688f;  // hd^-0.5
  for (int nt6 = 0; nt6 < 6; ++nt6) {
    const int n0 = (wave * 6 + nt6) * 16;    // output-column tile, 0..47
    v8f acc[4];
    #pragma unroll
    for (int mt = 0; mt < 4; ++mt) acc[mt] = (v8f){0.f,0.f,0.f,0.f,0.f,0.f,0.f,0.f};
    #pragma unroll
    for (int kc = 0; kc < 8; ++kc) {
      v16bf b = load_b(wqkvT + n0 * DIM + kc * 32, DIM, lane);
      #pragma unroll
      for (int mt = 0; mt < 4; ++mt) {
        v16bf a = load_a(xt + (mt * 16) * XLD + kc * 32, XLD, lane);
        acc[mt] = wmma_bf16(a, b, acc[mt]);
      }
    }
    const int   ncol = n0 + nn;
    const float bias = qkv_b[ncol];
    const int   part = n0 >> 8;              // 0=Q, 1=K, 2=V (uniform per wave-iter)
    #pragma unroll
    for (int mt = 0; mt < 4; ++mt) {
      #pragma unroll
      for (int i = 0; i < 8; ++i) {
        int   tok = mt * 16 + i + hi8;
        float val = acc[mt][i] + bias;
        if (part == 0)      qs[tok * QLD + ncol]         = f2bfu(val * scale);
        else if (part == 1) ks[tok * QLD + (ncol - 256)] = f2bfu(val);
        else                vt[(ncol - 512) * VLD + tok] = f2bfu(val);
      }
    }
  }
  __syncthreads();

  // ---- phase 3: attention, one head per wave (head-private LDS slices) ----
  const int h = wave;
  {
    v16bf qa[4], kb[4];
    #pragma unroll
    for (int t = 0; t < 4; ++t) {
      qa[t] = load_a(qs + (t * 16) * QLD + h * HD, QLD, lane);  // K = hd = 32
      kb[t] = load_b(ks + (t * 16) * QLD + h * HD, QLD, lane);
    }
    v8f s[4][4];
    #pragma unroll
    for (int mt = 0; mt < 4; ++mt)
      #pragma unroll
      for (int nt = 0; nt < 4; ++nt) {
        v8f z = (v8f){0.f,0.f,0.f,0.f,0.f,0.f,0.f,0.f};
        s[mt][nt] = wmma_bf16(qa[mt], kb[nt], z);
      }

    // + relative position bias (L2-resident, 128 KB)
    const float* rb = relb + h * SIZE * SIZE;
    #pragma unroll
    for (int mt = 0; mt < 4; ++mt)
      #pragma unroll
      for (int nt = 0; nt < 4; ++nt)
        #pragma unroll
        for (int i = 0; i < 8; ++i)
          s[mt][nt][i] += rb[(mt * 16 + i + hi8) * SIZE + nt * 16 + nn];

    // row softmax: row lives in 4 N-tiles x 16 lanes (within a 16-lane half)
    unsigned short* ph = probs + h * SIZE * PLD;
    #pragma unroll
    for (int mt = 0; mt < 4; ++mt) {
      #pragma unroll
      for (int i = 0; i < 8; ++i) {
        float x0 = s[mt][0][i], x1 = s[mt][1][i], x2 = s[mt][2][i], x3 = s[mt][3][i];
        float m = fmaxf(fmaxf(x0, x1), fmaxf(x2, x3));
        #pragma unroll
        for (int d = 8; d >= 1; d >>= 1) m = fmaxf(m, __shfl_xor(m, d, 32));
        x0 = __expf(x0 - m); x1 = __expf(x1 - m);
        x2 = __expf(x2 - m); x3 = __expf(x3 - m);
        float sum = x0 + x1 + x2 + x3;
        #pragma unroll
        for (int d = 8; d >= 1; d >>= 1) sum += __shfl_xor(sum, d, 32);
        float r = 1.0f / sum;
        unsigned short* pr = ph + (mt * 16 + i + hi8) * PLD;
        pr[ 0 + nn] = f2bfu(x0 * r);
        pr[16 + nn] = f2bfu(x1 * r);
        pr[32 + nn] = f2bfu(x2 * r);
        pr[48 + nn] = f2bfu(x3 * r);
      }
    }

    // out_h = P[64x64] @ V[64x32]  (B-frags read from transposed V)
    v8f o[4][2];
    #pragma unroll
    for (int mt = 0; mt < 4; ++mt)
      #pragma unroll
      for (int nt = 0; nt < 2; ++nt)
        o[mt][nt] = (v8f){0.f,0.f,0.f,0.f,0.f,0.f,0.f,0.f};
    #pragma unroll
    for (int mt = 0; mt < 4; ++mt) {
      #pragma unroll
      for (int kc = 0; kc < 2; ++kc) {
        v16bf a = load_a(ph + (mt * 16) * PLD + kc * 32, PLD, lane);
        #pragma unroll
        for (int nt = 0; nt < 2; ++nt) {
          v16bf b = load_b(vt + (h * HD + nt * 16) * VLD + kc * 32, VLD, lane);
          o[mt][nt] = wmma_bf16(a, b, o[mt][nt]);
        }
      }
    }
    // write head output into attn_out (reuses q region; columns are head-private)
    #pragma unroll
    for (int mt = 0; mt < 4; ++mt)
      #pragma unroll
      for (int nt = 0; nt < 2; ++nt)
        #pragma unroll
        for (int i = 0; i < 8; ++i)
          qs[(mt * 16 + i + hi8) * QLD + h * HD + nt * 16 + nn] = f2bfu(o[mt][nt][i]);
  }
  __syncthreads();

  // ---- phase 4: output projection  [64x256] @ proj_w[256x256] + bias -> global ----
  float* op = out + (size_t)win * SIZE * DIM;
  #pragma unroll
  for (int mt = 0; mt < 4; ++mt) {
    v16bf a[8];
    #pragma unroll
    for (int kc = 0; kc < 8; ++kc)
      a[kc] = load_a(qs + (mt * 16) * QLD + kc * 32, QLD, lane);
    #pragma unroll
    for (int nt2 = 0; nt2 < 2; ++nt2) {
      const int n0 = (wave * 2 + nt2) * 16;
      v8f acc = (v8f){0.f,0.f,0.f,0.f,0.f,0.f,0.f,0.f};
      #pragma unroll
      for (int kc = 0; kc < 8; ++kc) {
        v16bf b = load_b(projT + n0 * DIM + kc * 32, DIM, lane);
        acc = wmma_bf16(a[kc], b, acc);
      }
      const int   ncol = n0 + nn;
      const float pb   = proj_b[ncol];
      #pragma unroll
      for (int i = 0; i < 8; ++i)
        op[(size_t)(mt * 16 + i + hi8) * DIM + ncol] = acc[i] + pb;
    }
  }
}

// ---------------- launch ----------------
extern "C" void kernel_launch(void* const* d_in, const int* in_sizes, int n_in,
                              void* d_out, int out_size, void* d_ws, size_t ws_size,
                              hipStream_t stream) {
  const float* x      = (const float*)d_in[0];
  const float* qkv_w  = (const float*)d_in[1];  // [256 x 768]
  const float* qkv_b  = (const float*)d_in[2];  // [768]
  const float* proj_w = (const float*)d_in[3];  // [256 x 256]
  const float* proj_b = (const float*)d_in[4];  // [256]
  const float* btab   = (const float*)d_in[5];  // [343 x 8]
  float* out = (float*)d_out;

  // workspace: bf16 W^T (qkv, proj) + gathered rel-bias
  char* ws = (char*)d_ws;
  unsigned short* wqkvT = (unsigned short*)(ws);                            // 768*256 bf16
  unsigned short* projT = (unsigned short*)(ws + (size_t)QKV_N * DIM * 2);  // 256*256 bf16
  float*          relb  = (float*)(ws + (size_t)QKV_N * DIM * 2
                                      + (size_t)DIM * DIM * 2);             // 8*64*64 f32

  const int B = in_sizes[0] / (SIZE * DIM);

  prep_wT<<<(QKV_N * DIM + 255) / 256, 256, 0, stream>>>(qkv_w, wqkvT, DIM, QKV_N);
  prep_wT<<<(DIM * DIM + 255) / 256, 256, 0, stream>>>(proj_w, projT, DIM, DIM);
  prep_relbias<<<(NUM_HEADS * SIZE * SIZE + 255) / 256, 256, 0, stream>>>(btab, relb);

  (void)hipFuncSetAttribute((const void*)win_attn_fused,
                            hipFuncAttributeMaxDynamicSharedMemorySize, SMEM_BYTES);
  win_attn_fused<<<B, 256, SMEM_BYTES, stream>>>(x, qkv_b, proj_b, wqkvT, projT, relb, out);
}
